// GeometricMessagePassing_19275813224842
// MI455X (gfx1250) — matrix operand order repaired and checked by reading, not compiled, not run on hardware
//
#include <hip/hip_runtime.h>
#include <hip/hip_bf16.h>

#define NCH 128
#define BINS 64
#define CELLS (BINS * BINS)
#define TROWS 32

typedef __attribute__((ext_vector_type(16))) _Float16 v16h;
typedef __attribute__((ext_vector_type(8)))  float    v8f;

// ---------------------------------------------------------------- init
__global__ void gmp_init_k(float* __restrict__ grid, int* __restrict__ mm) {
    int idx = blockIdx.x * blockDim.x + threadIdx.x;
    if (idx < CELLS * NCH) grid[idx] = 0.0f;
    if (idx == 0) {
        mm[0] = 0x7F800000;  // min x (+inf bits; coords >= 0 so int order == float order)
        mm[1] = 0x7F800000;  // min y
        mm[2] = 0;           // max x
        mm[3] = 0;           // max y
    }
}

// ---------------------------------------------------------------- coord min/max
__global__ void gmp_minmax_k(const float* __restrict__ coords, int n, int* __restrict__ mm) {
    __shared__ float s0[256], s1[256], s2[256], s3[256];
    const int t = threadIdx.x;
    float mnx = __builtin_inff(), mny = __builtin_inff();
    float mxx = -__builtin_inff(), mxy = -__builtin_inff();
    for (int i = blockIdx.x * blockDim.x + t; i < n; i += gridDim.x * blockDim.x) {
        float cx = coords[2 * i + 0];
        float cy = coords[2 * i + 1];
        mnx = fminf(mnx, cx); mny = fminf(mny, cy);
        mxx = fmaxf(mxx, cx); mxy = fmaxf(mxy, cy);
    }
    s0[t] = mnx; s1[t] = mny; s2[t] = mxx; s3[t] = mxy;
    __syncthreads();
    for (int o = 128; o > 0; o >>= 1) {
        if (t < o) {
            s0[t] = fminf(s0[t], s0[t + o]); s1[t] = fminf(s1[t], s1[t + o]);
            s2[t] = fmaxf(s2[t], s2[t + o]); s3[t] = fmaxf(s3[t], s3[t + o]);
        }
        __syncthreads();
    }
    if (t == 0) {
        atomicMin(&mm[0], __float_as_int(s0[0]));
        atomicMin(&mm[1], __float_as_int(s1[0]));
        atomicMax(&mm[2], __float_as_int(s2[0]));
        atomicMax(&mm[3], __float_as_int(s3[0]));
    }
}

// ---------------------------------------------------------------- pw_w -> Bt f16 (transposed [n][k])
__global__ void gmp_cvtpw_k(const float* __restrict__ pw, _Float16* __restrict__ bt) {
    int idx = blockIdx.x * blockDim.x + threadIdx.x;   // 0..16383
    int k = idx >> 7, n = idx & 127;
    bt[n * NCH + k] = (_Float16)pw[k * NCH + n];
}

// ---------------------------------------------------------------- binning helper
__device__ __forceinline__ int gmp_bin(float c, float mn, float span) {
    int g = (int)floorf((c - mn) / span * 63.0f);
    return min(max(g, 0), 63);
}

// ---------------------------------------------------------------- scatter-add x into grid
__global__ void gmp_scatter_k(const float* __restrict__ x, const float* __restrict__ coords,
                              const int* __restrict__ mm, float* __restrict__ grid, int n) {
    const int t = threadIdx.x;   // 0..127, one channel each
    const float mnx = __int_as_float(mm[0]);
    const float mny = __int_as_float(mm[1]);
    const float spx = fmaxf(__int_as_float(mm[2]) - mnx, 1e-6f);
    const float spy = fmaxf(__int_as_float(mm[3]) - mny, 1e-6f);
    const int base = blockIdx.x * 128;
    for (int pp = 0; pp < 128; ++pp) {
        int p = base + pp;
        if (p >= n) break;
        float cx = coords[2 * p + 0];
        float cy = coords[2 * p + 1];
        int fid = gmp_bin(cx, mnx, spx) * BINS + gmp_bin(cy, mny, spy);
        atomicAdd(&grid[(size_t)fid * NCH + t], x[(size_t)p * NCH + t]);
    }
}

// ---------------------------------------------------------------- depthwise 3x3 conv + bias -> f16
__global__ void gmp_conv_k(const float* __restrict__ grid, const float* __restrict__ dwk,
                           const float* __restrict__ dwb, _Float16* __restrict__ convh) {
    const int i = blockIdx.x, j = blockIdx.y, c = threadIdx.x;
    float acc = dwb[c];
    #pragma unroll
    for (int di = -1; di <= 1; ++di) {
        int ii = i + di;
        if (ii < 0 || ii >= BINS) continue;
        #pragma unroll
        for (int dj = -1; dj <= 1; ++dj) {
            int jj = j + dj;
            if (jj < 0 || jj >= BINS) continue;
            acc += grid[((size_t)ii * BINS + jj) * NCH + c] *
                   dwk[((di + 1) * 3 + (dj + 1)) * NCH + c];
        }
    }
    convh[((size_t)i * BINS + j) * NCH + c] = (_Float16)acc;
}

// ---------------------------------------------------------------- persistent gather + WMMA GEMM + LN + residual
__global__ void __launch_bounds__(128)
gmp_gemm_ln_k(const _Float16* __restrict__ convh, const _Float16* __restrict__ bt,
              const float* __restrict__ coords, const int* __restrict__ mm,
              const float* __restrict__ x, const float* __restrict__ pwb,
              const float* __restrict__ lns, const float* __restrict__ lno,
              float* __restrict__ out, int n, int ntiles) {
    __shared__ _Float16 sB[NCH * NCH];            // 32 KB : Bt[n][k] f16 (staged once)
    __shared__ _Float16 sA[2][TROWS * NCH];       // 2 x 8 KB : double-buffered gathered rows
    __shared__ float    sC[TROWS * NCH];          // 16 KB : GEMM result + pw_b
    __shared__ float    sSum[4][TROWS], sSq[4][TROWS];
    __shared__ float    sMu[TROWS], sRs[TROWS];

    const int t = threadIdx.x;

    // ---- stage B once via CDNA5 async global->LDS DMA (ASYNCcnt tracked) ----
    {
        unsigned ldsB = (unsigned)(size_t)(void*)sB;   // LDS byte offset
        #pragma unroll 4
        for (int i = t; i < 2048; i += 128) {          // 2048 x 16 B = 32 KB
            unsigned loff = ldsB + (unsigned)i * 16u;
            unsigned goff = (unsigned)i * 16u;
            asm volatile("global_load_async_to_lds_b128 %0, %1, %2"
                         :: "v"(loff), "v"(goff), "s"(bt)
                         : "memory");
        }
        asm volatile("s_wait_asynccnt 0" ::: "memory");
    }
    __syncthreads();

    const float mnx = __int_as_float(mm[0]);
    const float mny = __int_as_float(mm[1]);
    const float spx = fmaxf(__int_as_float(mm[2]) - mnx, 1e-6f);
    const float spy = fmaxf(__int_as_float(mm[3]) - mny, 1e-6f);

    const int wave = t >> 5;
    const int lane = t & 31;
    const int lrow = lane & 15;
    const int lhi  = lane >> 4;
    const int mt     = wave >> 1;        // wave's m-tile (0..1)
    const int ntBase = (wave & 1) * 4;   // wave's n-tile group (0 or 4)

    // ---- hoist ALL B fragments into registers (16 x v16h = 128 VGPRs, loop-invariant) ----
    v16h bfrag[4][4];
    #pragma unroll
    for (int q = 0; q < 4; ++q) {
        #pragma unroll
        for (int kt = 0; kt < 4; ++kt) {
            const _Float16* bp = sB + ((ntBase + q) * 16 + lrow) * NCH + kt * 32 + lhi * 8;
            ((uint4*)&bfrag[q][kt])[0] = *(const uint4*)(bp);
            ((uint4*)&bfrag[q][kt])[1] = *(const uint4*)(bp + 16);
        }
    }

    const unsigned ldsA0 = (unsigned)(size_t)(void*)&sA[0][0];
    const unsigned ldsA1 = (unsigned)(size_t)(void*)&sA[1][0];
    const int arow = t >> 2, ach = t & 3;   // 4 threads/row, 64 B each

    // issue async gather of one tile's A rows into buffer `buf`
    auto stageA = [&](int tile, int buf) {
        int p = tile * TROWS + arow;
        unsigned ldsOff = (buf ? ldsA1 : ldsA0) + (unsigned)(arow * 256 + ach * 64);
        if (p < n) {
            float cx = coords[2 * p + 0];
            float cy = coords[2 * p + 1];
            int fid = gmp_bin(cx, mnx, spx) * BINS + gmp_bin(cy, mny, spy);
            unsigned goff = (unsigned)fid * 256u + (unsigned)ach * 64u;
            #pragma unroll
            for (int j = 0; j < 4; ++j) {
                asm volatile("global_load_async_to_lds_b128 %0, %1, %2"
                             :: "v"(ldsOff + (unsigned)j * 16u),
                                "v"(goff + (unsigned)j * 16u),
                                "s"(convh)
                             : "memory");
            }
        } else {
            uint4 z = make_uint4(0u, 0u, 0u, 0u);
            uint4* dst = (uint4*)(&sA[buf][0] + arow * NCH) + ach * 4;
            dst[0] = z; dst[1] = z; dst[2] = z; dst[3] = z;
        }
    };

    int tile = blockIdx.x;
    if (tile < ntiles) stageA(tile, 0);                    // prologue

    for (int it = 0; tile < ntiles; tile += gridDim.x, ++it) {
        const int buf  = it & 1;
        const int base = tile * TROWS;

        asm volatile("s_wait_asynccnt 0" ::: "memory");    // this wave's async A done
        __syncthreads();                                   // all waves' A in sA[buf]

        const int nextTile = tile + gridDim.x;
        if (nextTile < ntiles) stageA(nextTile, buf ^ 1);  // overlap gather with compute

        __builtin_prefetch(x + (size_t)base * NCH, 0, 0);  // residual rows

        // ---- A fragments for this wave's m-tile (ISA 16-bit A 16x32 layout) ----
        const _Float16* aBuf = &sA[buf][0];
        v16h afrag[4];
        #pragma unroll
        for (int kt = 0; kt < 4; ++kt) {
            const _Float16* ap = aBuf + (mt * 16 + lrow) * NCH + kt * 32 + lhi * 8;
            ((uint4*)&afrag[kt])[0] = *(const uint4*)(ap);
            ((uint4*)&afrag[kt])[1] = *(const uint4*)(ap + 16);
        }

        // ---- 16 register-fed WMMAs ----
        #pragma unroll
        for (int q = 0; q < 4; ++q) {
            v8f acc = {};
            #pragma unroll
            for (int kt = 0; kt < 4; ++kt) {
                acc = __builtin_amdgcn_wmma_f32_16x16x32_f16(
                    false, afrag[kt], false, bfrag[q][kt], (short)0, acc, false, false);
            }
            #pragma unroll
            for (int r = 0; r < 8; ++r) {
                int m   = mt * 16 + r + 8 * lhi;
                int col = (ntBase + q) * 16 + lrow;
                sC[m * NCH + col] = acc[r] + pwb[col];
            }
        }
        __syncthreads();

        // ---- LayerNorm stats: 4 segments x 32 rows ----
        {
            int row = t & 31, seg = t >> 5;
            float s = 0.0f, sq = 0.0f;
            #pragma unroll
            for (int c = 0; c < 32; ++c) {
                float v = sC[row * NCH + seg * 32 + c];
                s += v; sq += v * v;
            }
            sSum[seg][row] = s;
            sSq[seg][row]  = sq;
        }
        __syncthreads();
        if (t < TROWS) {
            float s = 0.0f, sq = 0.0f;
            #pragma unroll
            for (int seg = 0; seg < 4; ++seg) { s += sSum[seg][t]; sq += sSq[seg][t]; }
            float mu  = s * (1.0f / 128.0f);
            float var = sq * (1.0f / 128.0f) - mu * mu;
            sMu[t] = mu;
            sRs[t] = rsqrtf(var + 1e-5f);
        }
        __syncthreads();

        // ---- normalize + residual, coalesced float4 stores (8 per thread) ----
        {
            int row = t >> 2, c0 = (t & 3) * 32;
            int p = base + row;
            if (p < n) {
                float mu = sMu[row], rs = sRs[row];
                const float4* xr   = (const float4*)(x   + (size_t)p * NCH + c0);
                float4*       orow = (float4*)      (out + (size_t)p * NCH + c0);
                #pragma unroll
                for (int q = 0; q < 8; ++q) {
                    float4 xv = xr[q];
                    int c = c0 + q * 4;
                    float4 ov;
                    ov.x = xv.x + (sC[row * NCH + c + 0] - mu) * rs * lns[c + 0] + lno[c + 0];
                    ov.y = xv.y + (sC[row * NCH + c + 1] - mu) * rs * lns[c + 1] + lno[c + 1];
                    ov.z = xv.z + (sC[row * NCH + c + 2] - mu) * rs * lns[c + 2] + lno[c + 2];
                    ov.w = xv.w + (sC[row * NCH + c + 3] - mu) * rs * lns[c + 3] + lno[c + 3];
                    orow[q] = ov;
                }
            }
        }
        // sC/sA[buf] reuse is protected by next iteration's wait + barrier
    }
}

// ---------------------------------------------------------------- launcher
extern "C" void kernel_launch(void* const* d_in, const int* in_sizes, int n_in,
                              void* d_out, int out_size, void* d_ws, size_t ws_size,
                              hipStream_t stream) {
    const float* x      = (const float*)d_in[0];
    const float* coords = (const float*)d_in[1];
    const float* dwk    = (const float*)d_in[2];
    const float* dwb    = (const float*)d_in[3];
    const float* pww    = (const float*)d_in[4];
    const float* pwb    = (const float*)d_in[5];
    const float* lns    = (const float*)d_in[6];
    const float* lno    = (const float*)d_in[7];
    float*       out    = (float*)d_out;
    const int n = in_sizes[0] / NCH;

    char* ws = (char*)d_ws;
    float*     grid  = (float*)ws;                                     // 2 MB
    _Float16*  convh = (_Float16*)(ws + (size_t)CELLS * NCH * 4);      // 1 MB
    _Float16*  bt    = (_Float16*)(ws + 3u * 1024u * 1024u);           // 32 KB
    int*       mm    = (int*)(ws + 3u * 1024u * 1024u + 40u * 1024u);  // 16 B

    gmp_init_k<<<1024, 512, 0, stream>>>(grid, mm);
    gmp_minmax_k<<<512, 256, 0, stream>>>(coords, n, mm);
    gmp_cvtpw_k<<<64, 256, 0, stream>>>(pww, bt);
    gmp_scatter_k<<<(n + 127) / 128, 128, 0, stream>>>(x, coords, mm, grid, n);
    dim3 cgrid(BINS, BINS);
    gmp_conv_k<<<cgrid, NCH, 0, stream>>>(grid, dwk, dwb, convh);

    const int ntiles = (n + TROWS - 1) / TROWS;
    int nblocks = ntiles < 3072 ? ntiles : 3072;
    gmp_gemm_ln_k<<<nblocks, 128, 0, stream>>>(convh, bt, coords, mm,
                                               x, pwb, lns, lno, out, n, ntiles);
}
